// MultiHeadAttention_62414464746092
// MI455X (gfx1250) — matrix operand order repaired and checked by reading, not compiled, not run on hardware
//
#include <hip/hip_runtime.h>
#include <hip/hip_bf16.h>

typedef __attribute__((ext_vector_type(16))) _Float16 v16h;
typedef __attribute__((ext_vector_type(8)))  _Float16 v8h;
typedef __attribute__((ext_vector_type(8)))  float    v8f;
typedef __attribute__((ext_vector_type(4)))  float    v4f;
typedef __attribute__((ext_vector_type(4)))  int      v4i;

#define DIMX   1024
#define NHEAD  16
#define HID    64
#define BATCH  4
#define SEQ    2048
#define SROWS  (BATCH * SEQ)   // 8192

#if __has_builtin(__builtin_amdgcn_global_load_async_to_lds_b128) && \
    __has_builtin(__builtin_amdgcn_s_wait_asynccnt)
#define USE_ASYNC_LDS 1
typedef __attribute__((address_space(1))) v4i gv4i;   // global 128-bit chunk
typedef __attribute__((address_space(3))) v4i lv4i;   // LDS 128-bit chunk
#else
#define USE_ASYNC_LDS 0
#endif

static __device__ __forceinline__ v16h mk16(v8h lo, v8h hi) {
  union { v16h v; v8h h[2]; } u;
  u.h[0] = lo; u.h[1] = hi;
  return u.v;
}

static __device__ __forceinline__ v8h scale8(v8h x, _Float16 s) {
  for (int i = 0; i < 8; ++i) x[i] *= s;
  return x;
}

static __device__ __forceinline__ v8f wmma16(v16h a, v16h b, v8f c) {
  // v_wmma_f32_16x16x32_f16: D = A(16x32 f16) * B(32x16 f16) + C(16x16 f32)
  return __builtin_amdgcn_wmma_f32_16x16x32_f16(
      /*neg_a=*/false, a, /*neg_b=*/false, b,
      /*c_mod=*/(short)0, c, /*reuse_a=*/false, /*reuse_b=*/false);
}

// ---------------------------------------------------------------------------
// Generic tiled GEMM + bias:  Out[M,N] = A[M,K] * W[K,N] + bias[N]
// A f32 (convert) or f16 (async-copy) in global; W/bias f32; Out f16 or f32.
// OUT_TR: store f16 output transposed per head: Vt[(b*1024 + col)*SEQ + p]
// (8 acc elements per lane are consecutive p -> one packed v8h store).
// Block: 256 threads = 8 waves; block tile 32(M) x 64(N); K-step 32.
// ---------------------------------------------------------------------------
template <bool A_F32, bool OUT_F16, bool OUT_TR>
__global__ void __launch_bounds__(256)
gemm_bias_wmma(const void* __restrict__ Ap, const float* __restrict__ W,
               const float* __restrict__ bias, void* __restrict__ Outp,
               int M, int N, int K)
{
  __shared__ __align__(16) _Float16 aT[32][40];  // [m][k], row stride 80B
  __shared__ __align__(16) _Float16 bT[64][40];  // [n][k], row stride 80B

  const int t    = threadIdx.x;
  const int w    = t >> 5;
  const int lane = t & 31;
  const int ln   = lane & 15;
  const bool hi  = lane >= 16;
  const int tm   = w >> 2;       // 0..1
  const int tn   = w & 3;        // 0..3
  const int row0 = blockIdx.x * 32;
  const int col0 = blockIdx.y * 64;

  v8f acc = {};

  for (int k0 = 0; k0 < K; k0 += 32) {
    // Stage A 32x32 tile
    if (A_F32) {
      for (int i = 0; i < 4; ++i) {
        int e = t * 4 + i;
        int r = e >> 5, c = e & 31;
        aT[r][c] = (_Float16)((const float*)Ap)[(size_t)(row0 + r) * K + (k0 + c)];
      }
    } else {
#if USE_ASYNC_LDS
      if (t < 128) {            // 128 x 16B chunks cover 32x32 f16
        int r = t >> 2, c8 = (t & 3) * 8;
        __builtin_amdgcn_global_load_async_to_lds_b128(
            (gv4i*)((const _Float16*)Ap + (size_t)(row0 + r) * K + k0 + c8),
            (lv4i*)&aT[r][c8], 0, 0);
      }
#else
      if (t < 128) {
        int r = t >> 2, c8 = (t & 3) * 8;
        *(v8h*)&aT[r][c8] =
            *(const v8h*)((const _Float16*)Ap + (size_t)(row0 + r) * K + k0 + c8);
      }
#endif
    }
    // Stage W 32x64 tile transposed (convert to f16): bT[n][k]
    for (int i = 0; i < 8; ++i) {
      int e  = t * 8 + i;
      int kk = e >> 6, n = e & 63;
      bT[n][kk] = (_Float16)W[(size_t)(k0 + kk) * N + (col0 + n)];
    }
    if (k0 + 32 < K)
      __builtin_prefetch(&W[(size_t)(k0 + 32) * N + col0 + ln], 0, 0);
#if USE_ASYNC_LDS
    if (!A_F32) __builtin_amdgcn_s_wait_asynccnt(0);
#endif
    __syncthreads();

    // A fragment: lane = row (M), 16 contiguous K per half-wave
    const int ar   = tm * 16 + ln;
    const int aoff = hi ? 16 : 0;
    v16h afrag = mk16(*(const v8h*)&aT[ar][aoff],
                      *(const v8h*)&aT[ar][8 + aoff]);
    // B fragment: lane = col (N), 16 contiguous K rows per half-wave
    const int bn   = tn * 16 + ln;
    const int boff = hi ? 16 : 0;
    v16h bfrag = mk16(*(const v8h*)&bT[bn][boff],
                      *(const v8h*)&bT[bn][boff + 8]);

    acc = wmma16(afrag, bfrag, acc);
    __syncthreads();
  }

  const int col = col0 + tn * 16 + ln;
  const float bv = bias[col];
  if (OUT_TR) {
    // Transposed f16 store: 8 consecutive seq positions, one v8h store.
    union { v8h v; _Float16 e[8]; } o;
    for (int i = 0; i < 8; ++i) o.e[i] = (_Float16)(acc[i] + bv);
    const int bb = row0 / SEQ;
    const int p0 = (row0 % SEQ) + tm * 16 + (hi ? 8 : 0);
    *(v8h*)((_Float16*)Outp + ((size_t)bb * DIMX + col) * SEQ + p0) = o.v;
  } else {
    for (int i = 0; i < 8; ++i) {
      int row = row0 + tm * 16 + i + (hi ? 8 : 0);
      float v = acc[i] + bv;
      if (OUT_F16) ((_Float16*)Outp)[(size_t)row * N + col] = (_Float16)v;
      else         ((float*)Outp)[(size_t)row * N + col]    = v;
    }
  }
}

// ---------------------------------------------------------------------------
// Flash attention: grid (SEQ/128, B*H). Block = 8 waves; wave owns 16 queries.
// Streams over keys in steps of 32 with online softmax. All matmuls via WMMA.
// K tile [key][d] and pre-transposed V tile [d][key] are both staged with
// GLOBAL_LOAD_ASYNC_TO_LDS_B128, double-buffered so the DMA for tile n+1
// overlaps the WMMA/softmax of tile n. One barrier per iteration.
// Q/K are f16 [8192][1024]; Vt is f16 [(b*1024+h*64+d)][SEQ].
// ---------------------------------------------------------------------------
__global__ void __launch_bounds__(256)
flash_attn_wmma(const _Float16* __restrict__ Qh, const _Float16* __restrict__ Kh,
                const _Float16* __restrict__ Vt, _Float16* __restrict__ Ctx)
{
  __shared__ __align__(16) _Float16 ktile[2][32][72];   // [buf][key][d]
  __shared__ __align__(16) _Float16 vtileT[2][64][40];  // [buf][d][key]
  __shared__ __align__(16) float    stile[8][16][36];   // per-wave S transpose
  __shared__ __align__(16) float    alds[8][16];        // per-wave alpha / l stash

  const int t    = threadIdx.x;
  const int w    = t >> 5;
  const int lane = t & 31;
  const int ln   = lane & 15;
  const bool hi  = lane >= 16;
  const int b    = blockIdx.y >> 4;
  const int h    = blockIdx.y & 15;
  const int q0   = blockIdx.x * 128 + w * 16;   // wave's query base (within seq)

  const size_t vbase = ((size_t)b * DIMX + h * HID) * SEQ;  // Vt head base

  // Stage one 32-key tile (K row-major, Vt d-major) into buffer `buf`.
  auto stage = [&](int kb, int buf) {
    const int key = t >> 3, kd0 = (t & 7) * 8;            // 256 chunks for K
    const int d   = t >> 2, vc8 = (t & 3) * 8;            // 256 chunks for V
    const _Float16* ksrc = Kh + (size_t)(b * SEQ + kb + key) * DIMX + h * HID + kd0;
    const _Float16* vsrc = Vt + vbase + (size_t)d * SEQ + kb + vc8;
#if USE_ASYNC_LDS
    __builtin_amdgcn_global_load_async_to_lds_b128(
        (gv4i*)ksrc, (lv4i*)&ktile[buf][key][kd0], 0, 0);
    __builtin_amdgcn_global_load_async_to_lds_b128(
        (gv4i*)vsrc, (lv4i*)&vtileT[buf][d][vc8], 0, 0);
#else
    *(v8h*)&ktile[buf][key][kd0] = *(const v8h*)ksrc;
    *(v8h*)&vtileT[buf][d][vc8]  = *(const v8h*)vsrc;
#endif
  };

  // Load Q fragments (16 rows x 64 d) once; fold in the 1/sqrt(HID) scale.
  const _Float16* qbase = Qh + ((size_t)(b * SEQ + q0 + ln) * DIMX + h * HID);
  const int aoff = hi ? 16 : 0;
  const _Float16 qs = (_Float16)0.125f;
  v16h qf0 = mk16(scale8(*(const v8h*)(qbase + aoff), qs),
                  scale8(*(const v8h*)(qbase + 8 + aoff), qs));
  v16h qf1 = mk16(scale8(*(const v8h*)(qbase + 32 + aoff), qs),
                  scale8(*(const v8h*)(qbase + 40 + aoff), qs));

  v8f acc0 = {}, acc1 = {}, acc2 = {}, acc3 = {};
  float mprev = -1e30f;   // running row max (row = ln, duplicated in lane ln+16)
  float lsum  = 0.0f;     // running row sum

  stage(0, 0);            // prologue: tile 0 in flight

  int buf = 0;
  for (int kb = 0; kb < SEQ; kb += 32, buf ^= 1) {
#if USE_ASYNC_LDS
    __builtin_amdgcn_s_wait_asynccnt(0);
#endif
    __syncthreads();                       // tile `buf` ready; other buf free
    if (kb + 32 < SEQ) stage(kb + 32, buf ^ 1);   // overlap DMA with compute

    // ---- S(16x32) = Qs(16x64) @ K^T(64x32): two 16-key tiles, K-dim 32+32
    const int boff = hi ? 16 : 0;
    v8f s0 = {}, s1 = {};
    {
      const int key = ln;            // keys kb+0 .. kb+15
      v16h b0 = mk16(*(const v8h*)&ktile[buf][key][boff],
                     *(const v8h*)&ktile[buf][key][boff + 8]);
      v16h b1 = mk16(*(const v8h*)&ktile[buf][key][32 + boff],
                     *(const v8h*)&ktile[buf][key][32 + boff + 8]);
      s0 = wmma16(qf0, b0, s0);
      s0 = wmma16(qf1, b1, s0);
    }
    {
      const int key = 16 + ln;       // keys kb+16 .. kb+31
      v16h b0 = mk16(*(const v8h*)&ktile[buf][key][boff],
                     *(const v8h*)&ktile[buf][key][boff + 8]);
      v16h b1 = mk16(*(const v8h*)&ktile[buf][key][32 + boff],
                     *(const v8h*)&ktile[buf][key][32 + boff + 8]);
      s1 = wmma16(qf0, b0, s1);
      s1 = wmma16(qf1, b1, s1);
    }

    // ---- transpose raw S (C layout -> row-per-lane) through per-wave LDS
    float* sw = &stile[w][0][0];
    for (int i = 0; i < 8; ++i) {
      int r = hi ? (i + 8) : i;
      sw[r * 36 + ln]      = s0[i];
      sw[r * 36 + 16 + ln] = s1[i];
    }
    asm volatile("s_wait_dscnt 0x0" ::: "memory");

    // Lane ln (and ln+16) owns row ln: read its 16 values in A-fragment order.
    const float* srow = &stile[w][ln][0];
    v4f c0 = *(const v4f*)(srow + aoff);
    v4f c1 = *(const v4f*)(srow + aoff + 4);
    v4f c2 = *(const v4f*)(srow + 8 + aoff);
    v4f c3 = *(const v4f*)(srow + 8 + aoff + 4);
    float arr[16];
    for (int j = 0; j < 4; ++j) {
      arr[j] = c0[j]; arr[4 + j] = c1[j]; arr[8 + j] = c2[j]; arr[12 + j] = c3[j];
    }

    // ---- in-lane online softmax; one xor-16 shuffle each for max and sum
    float m16 = arr[0];
    for (int j = 1; j < 16; ++j) m16 = fmaxf(m16, arr[j]);
    float rowmax = fmaxf(m16, __shfl_xor(m16, 16, 32));
    float mn    = fmaxf(mprev, rowmax);
    float alpha = __expf(mprev - mn);
    mprev = mn;

    union { v16h v; _Float16 e[16]; } pu;
    float rs = 0.0f;
    for (int j = 0; j < 16; ++j) {
      float ev = __expf(arr[j] - mn);
      rs += ev;
      pu.e[j] = (_Float16)ev;
    }
    rs += __shfl_xor(rs, 16, 32);
    lsum = lsum * alpha + rs;

    // ---- redistribute alpha to the C layout (row i / i+8 per lane half)
    if (!hi) alds[w][ln] = alpha;
    asm volatile("s_wait_dscnt 0x0" ::: "memory");
    float ar8[8];
    for (int i = 0; i < 8; ++i) ar8[i] = alds[w][i + (hi ? 8 : 0)];
    for (int i = 0; i < 8; ++i) {
      acc0[i] *= ar8[i]; acc1[i] *= ar8[i]; acc2[i] *= ar8[i]; acc3[i] *= ar8[i];
    }

    // ---- acc(16x64) += P(16x32) @ V(32x64): four 16-col d-chunks
    {
      int d = ln;
      v16h vb = mk16(*(const v8h*)&vtileT[buf][d][boff],
                     *(const v8h*)&vtileT[buf][d][boff + 8]);
      acc0 = wmma16(pu.v, vb, acc0);
    }
    {
      int d = 16 + ln;
      v16h vb = mk16(*(const v8h*)&vtileT[buf][d][boff],
                     *(const v8h*)&vtileT[buf][d][boff + 8]);
      acc1 = wmma16(pu.v, vb, acc1);
    }
    {
      int d = 32 + ln;
      v16h vb = mk16(*(const v8h*)&vtileT[buf][d][boff],
                     *(const v8h*)&vtileT[buf][d][boff + 8]);
      acc2 = wmma16(pu.v, vb, acc2);
    }
    {
      int d = 48 + ln;
      v16h vb = mk16(*(const v8h*)&vtileT[buf][d][boff],
                     *(const v8h*)&vtileT[buf][d][boff + 8]);
      acc3 = wmma16(pu.v, vb, acc3);
    }
  }

  // ---- redistribute final l, normalize, store ctx (f16)
  if (!hi) alds[w][ln] = lsum;
  asm volatile("s_wait_dscnt 0x0" ::: "memory");
  for (int i = 0; i < 8; ++i) {
    float inv = 1.0f / alds[w][i + (hi ? 8 : 0)];
    int r = q0 + i + (hi ? 8 : 0);
    size_t rowoff = (size_t)(b * SEQ + r) * DIMX + h * HID;
    Ctx[rowoff + ln]      = (_Float16)(acc0[i] * inv);
    Ctx[rowoff + 16 + ln] = (_Float16)(acc1[i] * inv);
    Ctx[rowoff + 32 + ln] = (_Float16)(acc2[i] * inv);
    Ctx[rowoff + 48 + ln] = (_Float16)(acc3[i] * inv);
  }
}

// ---------------------------------------------------------------------------
extern "C" void kernel_launch(void* const* d_in, const int* in_sizes, int n_in,
                              void* d_out, int out_size, void* d_ws, size_t ws_size,
                              hipStream_t stream)
{
  const float* x  = (const float*)d_in[0];
  const float* Wq = (const float*)d_in[1];
  const float* bq = (const float*)d_in[2];
  const float* Wk = (const float*)d_in[3];
  const float* bk = (const float*)d_in[4];
  const float* Wv = (const float*)d_in[5];
  const float* bv = (const float*)d_in[6];
  const float* Wo = (const float*)d_in[7];
  const float* bo = (const float*)d_in[8];
  float* out = (float*)d_out;

  // Workspace: Q, K (row-major), Vt (transposed per head), Ctx. f16, 64 MB.
  _Float16* Qh = (_Float16*)d_ws;
  _Float16* Kh = Qh + (size_t)SROWS * DIMX;
  _Float16* Vt = Kh + (size_t)SROWS * DIMX;
  _Float16* Cx = Vt + (size_t)SROWS * DIMX;

  dim3 gGemm(SROWS / 32, DIMX / 64);
  gemm_bias_wmma<true, true, false><<<gGemm, 256, 0, stream>>>(x, Wq, bq, Qh, SROWS, DIMX, DIMX);
  gemm_bias_wmma<true, true, false><<<gGemm, 256, 0, stream>>>(x, Wk, bk, Kh, SROWS, DIMX, DIMX);
  gemm_bias_wmma<true, true, true ><<<gGemm, 256, 0, stream>>>(x, Wv, bv, Vt, SROWS, DIMX, DIMX);

  dim3 gAttn(SEQ / 128, BATCH * NHEAD);
  flash_attn_wmma<<<gAttn, 256, 0, stream>>>(Qh, Kh, Vt, Cx);

  gemm_bias_wmma<false, false, false><<<gGemm, 256, 0, stream>>>(Cx, Wo, bo, out, SROWS, DIMX, DIMX);
}
